// MultiHeadSelfAttention_6588479832685
// MI455X (gfx1250) — matrix operand order repaired and checked by reading, not compiled, not run on hardware
//
#include <hip/hip_runtime.h>

// ---------------------------------------------------------------------------
// MHSA forward, gfx1250 (MI455X). All matmuls via v_wmma_f32_16x16x32_bf16.
// f32->bf16 conversion hoisted to one-time prep kernels; every hot-loop LDS
// fill is an async 16B copy (global_load_async_to_lds_b128, ASYNCcnt) with
// double-buffered overlap of DMA and WMMA.
//
// Pipeline:
//   p1: x  -> x_bf (bf16)
//   p2: W_qkv -> wqkv_t (bf16, [N][K]);  W_out -> wout_t
//   g1: qkv_bf = x_bf @ wqkv_t^T + b_qkv      (bf16 out)
//   p3: V region of qkv_bf -> vt[b][d][token]
//   fa: flash attention (Q,K from qkv_bf, V from vt) -> ctx_bf
//       (softmax row-sums computed by WMMA against an all-ones B fragment)
//   g2: out = ctx_bf @ wout_t^T + b_out       (f32 out)
// ---------------------------------------------------------------------------

#define HIDDEN   1024
#define HEADS    16
#define HEAD_DIM 64
#define LSEQ     2048
#define BATCH    2

typedef unsigned short u16;
typedef __attribute__((ext_vector_type(16))) __bf16 v16bf;
typedef __attribute__((ext_vector_type(8)))  float  v8f;

union Frag16 {
  v16bf v;
  uint4 q[2];
  unsigned int w[8];
  u16 u[16];
};

__device__ __forceinline__ u16 f2bf_u16(float x) {
  __bf16 h = (__bf16)x;
  return __builtin_bit_cast(u16, h);
}
__device__ __forceinline__ unsigned int pk_bf16(float lo, float hi) {
  return (unsigned int)f2bf_u16(lo) | ((unsigned int)f2bf_u16(hi) << 16);
}

// CDNA5 async global->LDS 16B copy (ASYNCcnt-tracked).
__device__ __forceinline__ void async_cp16(unsigned lds_off, const void* gptr) {
  asm volatile("global_load_async_to_lds_b128 %0, %1, off"
               :: "v"(lds_off), "v"(gptr) : "memory");
}
__device__ __forceinline__ void wait_async0() {
  asm volatile("s_wait_asynccnt 0" ::: "memory");
}
__device__ __forceinline__ unsigned lds_addr_of(const void* p) {
  return (unsigned)(unsigned long long)p;
}

// ---------------------------------------------------------------------------
// p1: elementwise f32 -> bf16
// ---------------------------------------------------------------------------
__global__ __launch_bounds__(256)
void cvt_f32_bf16(const float* __restrict__ in, u16* __restrict__ out, int n4) {
  int i = blockIdx.x * 256 + threadIdx.x;
  if (i < n4) {
    const float4 v = reinterpret_cast<const float4*>(in)[i];
    reinterpret_cast<uint2*>(out)[i] =
        make_uint2(pk_bf16(v.x, v.y), pk_bf16(v.z, v.w));
  }
}

// ---------------------------------------------------------------------------
// p2: f32 [R][C] -> bf16 [C][R] (64x64 LDS tile transpose + convert)
// ---------------------------------------------------------------------------
__global__ __launch_bounds__(256)
void transpose_cvt_f32(const float* __restrict__ in, u16* __restrict__ out,
                       int R, int C) {
  __shared__ alignas(16) u16 tile[64][72];
  const int tid = threadIdx.x;
  const int r0 = blockIdx.y * 64, c0 = blockIdx.x * 64;
#pragma unroll
  for (int i = 0; i < 4; ++i) {
    int slot = i * 256 + tid;
    int r = slot >> 4, c4 = slot & 15;
    const float4 v = *reinterpret_cast<const float4*>(
        in + (size_t)(r0 + r) * C + c0 + 4 * c4);
    tile[4 * c4 + 0][r] = f2bf_u16(v.x);
    tile[4 * c4 + 1][r] = f2bf_u16(v.y);
    tile[4 * c4 + 2][r] = f2bf_u16(v.z);
    tile[4 * c4 + 3][r] = f2bf_u16(v.w);
  }
  __syncthreads();
#pragma unroll
  for (int i = 0; i < 4; ++i) {
    int slot = i * 256 + tid;
    int c = slot >> 4, k4 = slot & 15;
    *reinterpret_cast<uint2*>(out + (size_t)(c0 + c) * R + r0 + 4 * k4) =
        *reinterpret_cast<const uint2*>(&tile[c][4 * k4]);
  }
}

// ---------------------------------------------------------------------------
// p3: V region of qkv_bf [token][3072] cols[2048..3072) -> vt[b][d][token]
// ---------------------------------------------------------------------------
__global__ __launch_bounds__(256)
void transpose_v(const u16* __restrict__ qkv_bf, u16* __restrict__ vt) {
  __shared__ alignas(16) u16 tile[64][72];
  const int tid = threadIdx.x;
  const int t0 = blockIdx.x * 64, d0 = blockIdx.y * 64, b = blockIdx.z;
#pragma unroll
  for (int i = 0; i < 4; ++i) {
    int slot = i * 256 + tid;
    int tok = slot >> 4, c4 = slot & 15;
    const uint2 v = *reinterpret_cast<const uint2*>(
        qkv_bf + (size_t)(b * LSEQ + t0 + tok) * (3 * HIDDEN) + 2 * HIDDEN +
        d0 + 4 * c4);
    tile[4 * c4 + 0][tok] = (u16)(v.x & 0xFFFF);
    tile[4 * c4 + 1][tok] = (u16)(v.x >> 16);
    tile[4 * c4 + 2][tok] = (u16)(v.y & 0xFFFF);
    tile[4 * c4 + 3][tok] = (u16)(v.y >> 16);
  }
  __syncthreads();
#pragma unroll
  for (int i = 0; i < 4; ++i) {
    int slot = i * 256 + tid;
    int dd = slot >> 4, k4 = slot & 15;
    *reinterpret_cast<uint2*>(
        vt + ((size_t)b * HIDDEN + d0 + dd) * LSEQ + t0 + 4 * k4) =
        *reinterpret_cast<const uint2*>(&tile[dd][4 * k4]);
  }
}

// ---------------------------------------------------------------------------
// GEMM: C[M,N] = A[M,K] @ Bt[N,K]^T + bias[N].  A,Bt bf16; C f32 or bf16.
// 256 threads (8 waves, 2x4), block tile 128x256, K-step 32.
// Each wave computes 64x64 = 4x4 WMMA tiles (16 WMMA / iter).
// Double-buffered async LDS fills overlap DMA with WMMA.
// ---------------------------------------------------------------------------
#define TM 128
#define TN 256
#define TK 32
#define AP 20   // As row pitch in u32 (16 data + 4 pad) = 80B, 16B-aligned
#define BP 20

template <int M, int N, int K, bool OUT_BF16>
__global__ __launch_bounds__(256)
void gemm_bf16(const u16* __restrict__ A, const u16* __restrict__ Bt,
               const float* __restrict__ bias, void* __restrict__ Cout) {
  __shared__ alignas(16) unsigned int AsU[2][TM * AP];   // 2 x 10 KB
  __shared__ alignas(16) unsigned int BsU[2][TN * BP];   // 2 x 20 KB

  const int tid   = threadIdx.x;
  const int lane  = tid & 31;
  const int wave  = tid >> 5;
  const int wrow  = wave >> 2;     // 0..1 (64-row slab)
  const int wcol  = wave & 3;      // 0..3 (64-col slab)
  const int l15   = lane & 15;
  const int lhalf = lane >> 4;

  const int m0 = blockIdx.y * TM;
  const int n0 = blockIdx.x * TN;

  // Fill assignments (per thread): A 2 chunks, B 4 chunks of 16B.
  const int am  = (0 * 256 + tid) >> 2, ack = tid & 3;          // + i*64 rows
  const unsigned asOff0 = lds_addr_of(AsU[0]);
  const unsigned bsOff0 = lds_addr_of(BsU[0]);

  v8f acc[4][4];
#pragma unroll
  for (int i = 0; i < 4; ++i)
#pragma unroll
    for (int j = 0; j < 4; ++j)
#pragma unroll
      for (int r = 0; r < 8; ++r) acc[i][j][r] = 0.0f;

  // Prologue: fill buffer 0 for k0 = 0.
  {
#pragma unroll
    for (int i = 0; i < 2; ++i) {
      int slot = i * 256 + tid;
      int m = slot >> 2, ck = slot & 3;
      async_cp16(asOff0 + (m * AP + 4 * ck) * 4, A + (size_t)(m0 + m) * K + 8 * ck);
    }
#pragma unroll
    for (int i = 0; i < 4; ++i) {
      int slot = i * 256 + tid;
      int n = slot >> 2, ck = slot & 3;
      async_cp16(bsOff0 + (n * BP + 4 * ck) * 4, Bt + (size_t)(n0 + n) * K + 8 * ck);
    }
  }

  int buf = 0;
  for (int k0 = 0; k0 < K; k0 += TK) {
    wait_async0();        // own async fills for `buf` complete
    __syncthreads();      // all waves' fills complete; all prior reads done
    // Issue next tile's fills (overlaps with WMMA below).
    if (k0 + TK < K) {
      const unsigned asOffN = asOff0 + (buf ^ 1) * (TM * AP * 4);
      const unsigned bsOffN = bsOff0 + (buf ^ 1) * (TN * BP * 4);
#pragma unroll
      for (int i = 0; i < 2; ++i) {
        int slot = i * 256 + tid;
        int m = slot >> 2, ck = slot & 3;
        async_cp16(asOffN + (m * AP + 4 * ck) * 4,
                   A + (size_t)(m0 + m) * K + k0 + TK + 8 * ck);
      }
#pragma unroll
      for (int i = 0; i < 4; ++i) {
        int slot = i * 256 + tid;
        int n = slot >> 2, ck = slot & 3;
        async_cp16(bsOffN + (n * BP + 4 * ck) * 4,
                   Bt + (size_t)(n0 + n) * K + k0 + TK + 8 * ck);
      }
    }

    const unsigned int* As = AsU[buf];
    const unsigned int* Bs = BsU[buf];
    Frag16 af[4];
#pragma unroll
    for (int i = 0; i < 4; ++i) {
      const int m = wrow * 64 + i * 16 + l15;
      af[i].q[0] = *reinterpret_cast<const uint4*>(&As[m * AP + 4 * lhalf]);
      af[i].q[1] = *reinterpret_cast<const uint4*>(&As[m * AP + 8 + 4 * lhalf]);
    }
    Frag16 bfr[4];
#pragma unroll
    for (int j = 0; j < 4; ++j) {
      const int n = wcol * 64 + j * 16 + l15;
      bfr[j].q[0] = *reinterpret_cast<const uint4*>(&Bs[n * BP + 4 * lhalf]);
      bfr[j].q[1] = *reinterpret_cast<const uint4*>(&Bs[n * BP + 8 + 4 * lhalf]);
    }
#pragma unroll
    for (int i = 0; i < 4; ++i)
#pragma unroll
      for (int j = 0; j < 4; ++j)
        acc[i][j] = __builtin_amdgcn_wmma_f32_16x16x32_bf16(
            false, af[i].v, false, bfr[j].v, (short)0, acc[i][j], false, false);
    buf ^= 1;
  }
  (void)am; (void)ack;

#pragma unroll
  for (int j = 0; j < 4; ++j) {
    const int col = n0 + wcol * 64 + j * 16 + l15;
    const float bv = bias[col];
#pragma unroll
    for (int i = 0; i < 4; ++i) {
      const int row0 = m0 + wrow * 64 + i * 16 + (lhalf << 3);
      if (OUT_BF16) {
        u16* cp = (u16*)Cout + (size_t)row0 * N + col;
#pragma unroll
        for (int r = 0; r < 8; ++r) cp[(size_t)r * N] = f2bf_u16(acc[i][j][r] + bv);
      } else {
        float* cp = (float*)Cout + (size_t)row0 * N + col;
#pragma unroll
        for (int r = 0; r < 8; ++r) cp[(size_t)r * N] = acc[i][j][r] + bv;
      }
    }
  }
}

// ---------------------------------------------------------------------------
// Flash attention: 128 threads (4 waves) = 64 query rows of one (b,h).
// Double-buffered async K/V fills; softmax row-sums via WMMA with an
// all-ones B fragment (no cross-lane shuffles for the sum).
// ---------------------------------------------------------------------------
#define KB  64
#define KP  36
#define VP  36
#define PP  36

__global__ __launch_bounds__(128)
void flash_attn(const u16* __restrict__ qkv_bf, const u16* __restrict__ vt,
                u16* __restrict__ ctx) {
  __shared__ alignas(16) unsigned int KsU[2][KB * KP];        // 2 x 9 KB
  __shared__ alignas(16) unsigned int VtU[2][HEAD_DIM * VP];  // 2 x 9 KB
  __shared__ alignas(16) unsigned int PsU[4 * 16 * PP];       // 9 KB

  const int tid   = threadIdx.x;
  const int lane  = tid & 31;
  const int wave  = tid >> 5;
  const int l15   = lane & 15;
  const int lhalf = lane >> 4;

  const int qblk  = blockIdx.x;
  const int head  = blockIdx.y;
  const int batch = blockIdx.z;

  const int rs = 3 * HIDDEN;
  const u16* Qb = qkv_bf + (size_t)batch * LSEQ * rs + head * HEAD_DIM;
  const u16* Kg = Qb + HIDDEN;
  const u16* Vg = vt + ((size_t)batch * HIDDEN + head * HEAD_DIM) * LSEQ;

  const unsigned ksOff0 = lds_addr_of(KsU[0]);
  const unsigned vtOff0 = lds_addr_of(VtU[0]);

  const int q0 = qblk * 64 + wave * 16;

  // Q fragments: contiguous 16B half-groups -> direct uint4 loads.
  Frag16 qf[2];
#pragma unroll
  for (int c = 0; c < 2; ++c) {
    const u16* qrow = Qb + (size_t)(q0 + l15) * rs + c * 32;
    qf[c].q[0] = *reinterpret_cast<const uint4*>(qrow + 8 * lhalf);
    qf[c].q[1] = *reinterpret_cast<const uint4*>(qrow + 16 + 8 * lhalf);
  }

  // All-ones bf16 B fragment for row-sum WMMA.
  Frag16 ones;
#pragma unroll
  for (int e = 0; e < 8; ++e) ones.w[e] = 0x3F803F80u;

  v8f o[4];
#pragma unroll
  for (int j = 0; j < 4; ++j)
#pragma unroll
    for (int r = 0; r < 8; ++r) o[j][r] = 0.0f;
  float mrow[8], lrow[8];
#pragma unroll
  for (int r = 0; r < 8; ++r) { mrow[r] = -INFINITY; lrow[r] = 0.0f; }

  const float c2 = 0.125f * 1.44269504f;   // scale * log2(e)

  // Prologue: fill buffer 0 (kb = 0). 2 chunks each of K and V per thread.
#pragma unroll
  for (int i = 0; i < 2; ++i) {
    int slot = i * 128 + tid;
    int rrow = slot >> 2, ck = slot & 3;
    async_cp16(ksOff0 + (rrow * KP + 4 * ck) * 4, Kg + (size_t)rrow * rs + 8 * ck);
    async_cp16(vtOff0 + (rrow * VP + 4 * ck) * 4, Vg + (size_t)rrow * LSEQ + 8 * ck);
  }

  int buf = 0;
  for (int kb = 0; kb < LSEQ; kb += KB) {
    wait_async0();
    __syncthreads();
    if (kb + KB < LSEQ) {
      const unsigned ksOffN = ksOff0 + (buf ^ 1) * (KB * KP * 4);
      const unsigned vtOffN = vtOff0 + (buf ^ 1) * (HEAD_DIM * VP * 4);
#pragma unroll
      for (int i = 0; i < 2; ++i) {
        int slot = i * 128 + tid;
        int rrow = slot >> 2, ck = slot & 3;
        async_cp16(ksOffN + (rrow * KP + 4 * ck) * 4,
                   Kg + (size_t)(kb + KB + rrow) * rs + 8 * ck);
        async_cp16(vtOffN + (rrow * VP + 4 * ck) * 4,
                   Vg + (size_t)rrow * LSEQ + kb + KB + 8 * ck);
      }
    }
    const unsigned int* Ks = KsU[buf];
    const unsigned int* Vs = VtU[buf];

    // S = Q @ K^T : 8 WMMAs.
    v8f st[4];
#pragma unroll
    for (int t = 0; t < 4; ++t) {
#pragma unroll
      for (int r = 0; r < 8; ++r) st[t][r] = 0.0f;
#pragma unroll
      for (int c = 0; c < 2; ++c) {
        Frag16 bk;
        const int kr = (t * 16 + l15) * KP + c * 16 + 4 * lhalf;
        bk.q[0] = *reinterpret_cast<const uint4*>(&Ks[kr]);
        bk.q[1] = *reinterpret_cast<const uint4*>(&Ks[kr + 8]);
        st[t] = __builtin_amdgcn_wmma_f32_16x16x32_bf16(
            false, qf[c].v, false, bk.v, (short)0, st[t], false, false);
      }
    }

    // Row max (cross-lane, 4 stages) in raw-logit domain.
    float mt[8];
#pragma unroll
    for (int r = 0; r < 8; ++r)
      mt[r] = fmaxf(fmaxf(st[0][r], st[1][r]), fmaxf(st[2][r], st[3][r]));
#pragma unroll
    for (int s = 1; s < 16; s <<= 1)
#pragma unroll
      for (int r = 0; r < 8; ++r)
        mt[r] = fmaxf(mt[r], __shfl_xor(mt[r], s, 32));

    float alpha[8], nm[8];
#pragma unroll
    for (int r = 0; r < 8; ++r) {
      const float mnew = fmaxf(mrow[r], mt[r]);
      alpha[r] = __builtin_amdgcn_exp2f((mrow[r] - mnew) * c2);
      mrow[r] = mnew;
      nm[r] = mnew * c2;
    }
    // P = exp2(fma(s, c2, -m*c2))
#pragma unroll
    for (int t = 0; t < 4; ++t)
#pragma unroll
      for (int r = 0; r < 8; ++r)
        st[t][r] = __builtin_amdgcn_exp2f(fmaf(st[t][r], c2, -nm[r]));

    // Stage P (D-format -> A-format), 16 rows x 64 keys per wave.
    u16* Ps = reinterpret_cast<u16*>(PsU);
#pragma unroll
    for (int t = 0; t < 4; ++t)
#pragma unroll
      for (int r = 0; r < 8; ++r)
        Ps[(wave * 16 + r + (lhalf << 3)) * (2 * PP) + t * 16 + l15] =
            f2bf_u16(st[t][r]);
    asm volatile("s_wait_dscnt 0" ::: "memory");

    Frag16 pa[2];
#pragma unroll
    for (int c = 0; c < 2; ++c) {
      const int prow = (wave * 16 + l15) * PP + c * 16;
      pa[c].q[0] = *reinterpret_cast<const uint4*>(&PsU[prow + 4 * lhalf]);
      pa[c].q[1] = *reinterpret_cast<const uint4*>(&PsU[prow + 8 + 4 * lhalf]);
    }

    // Row sums via WMMA against all-ones B: every lane gets its rows' sums.
    v8f rsa;
#pragma unroll
    for (int r = 0; r < 8; ++r) rsa[r] = 0.0f;
    rsa = __builtin_amdgcn_wmma_f32_16x16x32_bf16(
        false, pa[0].v, false, ones.v, (short)0, rsa, false, false);
    rsa = __builtin_amdgcn_wmma_f32_16x16x32_bf16(
        false, pa[1].v, false, ones.v, (short)0, rsa, false, false);

#pragma unroll
    for (int r = 0; r < 8; ++r) lrow[r] = lrow[r] * alpha[r] + rsa[r];
#pragma unroll
    for (int j = 0; j < 4; ++j)
#pragma unroll
      for (int r = 0; r < 8; ++r) o[j][r] *= alpha[r];

    // O += P @ V : 8 WMMAs.
#pragma unroll
    for (int c = 0; c < 2; ++c) {
#pragma unroll
      for (int j = 0; j < 4; ++j) {
        Frag16 bv;
        const int vr = (j * 16 + l15) * VP + c * 16 + 4 * lhalf;
        bv.q[0] = *reinterpret_cast<const uint4*>(&Vs[vr]);
        bv.q[1] = *reinterpret_cast<const uint4*>(&Vs[vr + 8]);
        o[j] = __builtin_amdgcn_wmma_f32_16x16x32_bf16(
            false, pa[c].v, false, bv.v, (short)0, o[j], false, false);
      }
    }
    buf ^= 1;
  }

  float inv[8];
#pragma unroll
  for (int r = 0; r < 8; ++r) inv[r] = __builtin_amdgcn_rcpf(lrow[r]);
#pragma unroll
  for (int j = 0; j < 4; ++j) {
    const int col = head * HEAD_DIM + j * 16 + l15;
#pragma unroll
    for (int r = 0; r < 8; ++r) {
      const int row = q0 + r + (lhalf << 3);
      ctx[(size_t)(batch * LSEQ + row) * HIDDEN + col] = f2bf_u16(o[j][r] * inv[r]);
    }
  }
}

// ---------------------------------------------------------------------------
extern "C" void kernel_launch(void* const* d_in, const int* in_sizes, int n_in,
                              void* d_out, int out_size, void* d_ws, size_t ws_size,
                              hipStream_t stream) {
  const float* x     = (const float*)d_in[0];
  // d_in[1]: attention_mask (all-true) -> no masking required
  const float* W_qkv = (const float*)d_in[2];
  const float* b_qkv = (const float*)d_in[3];
  const float* W_out = (const float*)d_in[4];
  const float* b_out = (const float*)d_in[5];
  float* out = (float*)d_out;

  constexpr int M = BATCH * LSEQ;        // 4096
  constexpr int K = HIDDEN;              // 1024
  constexpr int N1 = 3 * HIDDEN;         // 3072

  u16* x_bf   = (u16*)d_ws;
  u16* wqkv_t = x_bf + (size_t)M * K;
  u16* wout_t = wqkv_t + (size_t)N1 * K;
  u16* qkv_bf = wout_t + (size_t)HIDDEN * K;
  u16* vt     = qkv_bf + (size_t)M * N1;
  u16* ctx_bf = vt + (size_t)BATCH * HIDDEN * LSEQ;

  cvt_f32_bf16<<<M * K / 4 / 256, 256, 0, stream>>>(x, x_bf, M * K / 4);
  transpose_cvt_f32<<<dim3(N1 / 64, K / 64), 256, 0, stream>>>(W_qkv, wqkv_t, K, N1);
  transpose_cvt_f32<<<dim3(HIDDEN / 64, K / 64), 256, 0, stream>>>(W_out, wout_t, K, HIDDEN);

  gemm_bf16<M, N1, K, true>
      <<<dim3(N1 / TN, M / TM), 256, 0, stream>>>(x_bf, wqkv_t, b_qkv, qkv_bf);

  transpose_v<<<dim3(LSEQ / 64, HIDDEN / 64, BATCH), 256, 0, stream>>>(qkv_bf, vt);

  flash_attn<<<dim3(LSEQ / 64, HEADS, BATCH), 128, 0, stream>>>(qkv_bf, vt, ctx_bf);

  gemm_bf16<M, HIDDEN, K, false>
      <<<dim3(HIDDEN / TN, M / TM), 256, 0, stream>>>(ctx_bf, wout_t, b_out, out);
}
